// CausalSelfAttention_25142738551000
// MI455X (gfx1250) — matrix operand order repaired and checked by reading, not compiled, not run on hardware
//
#include <hip/hip_runtime.h>
#include <hip/hip_bf16.h>
#include <stdint.h>

// ---------- types ----------
typedef _Float16 half_t;
typedef __attribute__((ext_vector_type(8)))  _Float16 v8h;
typedef __attribute__((ext_vector_type(16))) _Float16 v16h;
typedef __attribute__((ext_vector_type(8)))  float    v8f;

#define WMMA_F16(A, B, C) \
    __builtin_amdgcn_wmma_f32_16x16x32_f16(false, (A), false, (B), (short)0, (C), false, false)

__device__ __forceinline__ v16h hcat(v8h lo, v8h hi) {
    return __builtin_shufflevector(lo, hi, 0,1,2,3,4,5,6,7,8,9,10,11,12,13,14,15);
}

__device__ __forceinline__ float hw_max16(float v) {   // reduce across a 16-lane half
    v = fmaxf(v, __shfl_xor(v, 1));
    v = fmaxf(v, __shfl_xor(v, 2));
    v = fmaxf(v, __shfl_xor(v, 4));
    v = fmaxf(v, __shfl_xor(v, 8));
    return v;
}
__device__ __forceinline__ float hw_sum16(float v) {
    v += __shfl_xor(v, 1);
    v += __shfl_xor(v, 2);
    v += __shfl_xor(v, 4);
    v += __shfl_xor(v, 8);
    return v;
}

// Async copy: 16B per lane, global -> LDS, tracked by ASYNCcnt.
__device__ __forceinline__ void async_b128(uint32_t lds_addr, uint32_t voff, uint64_t sbase) {
    asm volatile("global_load_async_to_lds_b128 %0, %1, %2"
                 :: "v"(lds_addr), "v"(voff), "s"(sbase) : "memory");
}
__device__ __forceinline__ void wait_async0() {
    asm volatile("s_wait_asynccnt 0x0" ::: "memory");
}

// Problem constants
#define BB 2
#define TT 2048
#define CC 1024
#define HH 16
#define HD 64

// ============================================================================
// Kernel 0: bulk f32 -> f16 conversion (one pass; makes all GEMM staging a
// pure f16 copy so tiles can be moved with async-to-LDS DMA only).
// ============================================================================
__global__ __launch_bounds__(256)
void cvt_f16_kernel(const float* __restrict__ src, half_t* __restrict__ dst, int n8)
{
    int i = blockIdx.x * 256 + threadIdx.x;
    if (i >= n8) return;
    const float4* s = (const float4*)(src) + (size_t)i * 2;
    float4 f0 = s[0], f1 = s[1];
    v8h h;
    h[0] = (half_t)f0.x; h[1] = (half_t)f0.y; h[2] = (half_t)f0.z; h[3] = (half_t)f0.w;
    h[4] = (half_t)f1.x; h[5] = (half_t)f1.y; h[6] = (half_t)f1.z; h[7] = (half_t)f1.w;
    *((v8h*)dst + i) = h;
}

// ============================================================================
// Kernel 1: QKV projection.  Y(4096 x 3072) = Xh(4096 x 1024) @ Wh^T (all f16).
// 128x64 block tile, 4 waves x 32x64, K-step 32.  Double-buffered LDS with
// async-to-LDS DMA staging for both A and B tiles.
// Epilogue: Q (scaled 1/8) and K in (B,H,T,hd); V TRANSPOSED as (B,H,hd,T)
// (lane's 8 C-rows are contiguous t -> one v8h store).
// ============================================================================
#define SA 56   // LDS row stride in halfs (112B: 16B-aligned, conflict-free)

__global__ __launch_bounds__(128)
void qkv_gemm_kernel(const half_t* __restrict__ Xh, const half_t* __restrict__ Wh,
                     half_t* __restrict__ Qh, half_t* __restrict__ Kh,
                     half_t* __restrict__ Vt)
{
    __shared__ half_t As[2][128 * SA];
    __shared__ half_t Bs[2][64 * SA];
    const int tid  = threadIdx.x;
    const int wave = tid >> 5;
    const int lane = tid & 31;
    const int hiH  = (lane >> 4) & 1;
    const int lr   = lane & 15;
    const int m0   = blockIdx.y * 128;
    const int n0   = blockIdx.x * 64;

    auto stage = [&](int k0, int buf) {
        const uint64_t abase = (uint64_t)(uintptr_t)(Xh + (size_t)m0 * CC + k0);
#pragma unroll
        for (int i = 0; i < 4; ++i) {          // A: 128x32 halfs = 8KB
            const int lin = ((wave * 4 + i) * 32 + lane) * 8;
            const int r = lin >> 5, c = lin & 31;
            async_b128((uint32_t)(uintptr_t)&As[buf][r * SA + c],
                       (uint32_t)((r * CC + c) * 2), abase);
        }
        const uint64_t bbase = (uint64_t)(uintptr_t)(Wh + (size_t)n0 * CC + k0);
#pragma unroll
        for (int i = 0; i < 2; ++i) {          // B: 64x32 halfs = 4KB
            const int lin = ((wave * 2 + i) * 32 + lane) * 8;
            const int r = lin >> 5, c = lin & 31;
            async_b128((uint32_t)(uintptr_t)&Bs[buf][r * SA + c],
                       (uint32_t)((r * CC + c) * 2), bbase);
        }
    };

    v8f acc[2][4] = {};
    stage(0, 0);

    for (int ki = 0; ki < CC / 32; ++ki) {
        wait_async0();
        __syncthreads();
        if (ki + 1 < CC / 32) stage((ki + 1) * 32, (ki + 1) & 1);

        const half_t* Ab = As[ki & 1];
        const half_t* Bb = Bs[ki & 1];
        const int ko = hiH ? 8 : 0;
        const half_t* ar0 = &Ab[(wave * 16 + lr) * SA];
        const half_t* ar1 = &Ab[(64 + wave * 16 + lr) * SA];
        v16h a0 = hcat(*(const v8h*)(ar0 + ko), *(const v8h*)(ar0 + 16 + ko));
        v16h a1 = hcat(*(const v8h*)(ar1 + ko), *(const v8h*)(ar1 + 16 + ko));
        const int kb = hiH ? 16 : 0;
#pragma unroll
        for (int t = 0; t < 4; ++t) {
            const half_t* br = &Bb[(t * 16 + lr) * SA];
            v16h bfrag = hcat(*(const v8h*)(br + ko), *(const v8h*)(br + 16 + ko));
            acc[0][t] = WMMA_F16(a0, bfrag, acc[0][t]);
            acc[1][t] = WMMA_F16(a1, bfrag, acc[1][t]);
        }
    }

#pragma unroll
    for (int t = 0; t < 4; ++t) {
        const int n     = n0 + t * 16 + lr;
        const int which = n >> 10;          // 0=q 1=k 2=v
        const int c     = n & (CC - 1);
        const int h     = c >> 6;
        const int dd    = c & 63;
#pragma unroll
        for (int s = 0; s < 2; ++s) {
            const int mb  = m0 + s * 64 + wave * 16 + hiH * 8;  // first of 8 rows
            const int b   = mb >> 11;
            const int tr0 = mb & (TT - 1);
            if (which == 2) {
                v8h pk;
#pragma unroll
                for (int j = 0; j < 8; ++j) pk[j] = (half_t)acc[s][t][j];
                size_t off = ((size_t)(b * HH + h) * HD + dd) * TT + tr0;
                *(v8h*)(Vt + off) = pk;
            } else {
                half_t* dst  = (which == 0) ? Qh : Kh;
                const float sc = (which == 0) ? 0.125f : 1.0f;  // 1/sqrt(hd) in Q
#pragma unroll
                for (int j = 0; j < 8; ++j) {
                    size_t off = ((size_t)(b * HH + h) * TT + (tr0 + j)) * HD + dd;
                    dst[off] = (half_t)(acc[s][t][j] * sc);
                }
            }
        }
    }
}

// ============================================================================
// Kernel 2: causal flash attention.  One block = (b,h) x 64 q-rows; 4 waves x
// 16 q-rows.  K tile [key][d] and V^T tile [d][key] double-buffered and staged
// with async-to-LDS DMA (padded strides via per-lane addresses).
// ============================================================================
#define SK 72   // LDS row stride in halfs (144B: 16B-aligned, conflict-free)

__global__ __launch_bounds__(128)
void attn_kernel(const half_t* __restrict__ Qh, const half_t* __restrict__ Kh,
                 const half_t* __restrict__ Vt, half_t* __restrict__ Ah)
{
    __shared__ half_t Ks[2][64 * SK];       // [key][d]
    __shared__ half_t Vs[2][64 * SK];       // [d][key]
    __shared__ half_t Ps[4][16 * SK];       // per-wave P tile [M][key]

    const int tid   = threadIdx.x;
    const int wave  = tid >> 5;
    const int lane  = tid & 31;
    const int hiH   = (lane >> 4) & 1;
    const int lr    = lane & 15;
    const int qtile = blockIdx.x;           // 0..31
    const int bh    = blockIdx.y;           // b*16 + h
    const size_t base = (size_t)bh * TT * HD;

    auto stageKV = [&](int jt, int buf) {
        const uint64_t kbase = (uint64_t)(uintptr_t)(Kh + base + (size_t)jt * 64 * HD);
        const uint64_t vbase = (uint64_t)(uintptr_t)(Vt + base + (size_t)jt * 64);
#pragma unroll
        for (int i = 0; i < 4; ++i) {
            const int lin = ((wave * 4 + i) * 32 + lane) * 8;  // half index
            const int r = lin >> 6, c = lin & 63;
            async_b128((uint32_t)(uintptr_t)&Ks[buf][r * SK + c],
                       (uint32_t)(lin * 2), kbase);
            async_b128((uint32_t)(uintptr_t)&Vs[buf][r * SK + c],
                       (uint32_t)((r * TT + c) * 2), vbase);
        }
    };

    // Q A-fragments straight from global (rows contiguous, 16B-aligned)
    const int qrow = qtile * 64 + wave * 16 + lr;
    const half_t* qp = Qh + base + (size_t)qrow * HD;
    const int ko = hiH ? 8 : 0;
    v16h aq0 = hcat(*(const v8h*)(qp + ko),      *(const v8h*)(qp + 16 + ko));
    v16h aq1 = hcat(*(const v8h*)(qp + 32 + ko), *(const v8h*)(qp + 48 + ko));

    float m_i[8], l_i[8];
#pragma unroll
    for (int j = 0; j < 8; ++j) { m_i[j] = -1e30f; l_i[j] = 0.0f; }
    v8f o[4] = {};

    stageKV(0, 0);

    for (int jt = 0; jt <= qtile; ++jt) {
        wait_async0();
        __syncthreads();
        if (jt < qtile) stageKV(jt + 1, (jt + 1) & 1);

        const half_t* Kb = Ks[jt & 1];
        const half_t* Vb = Vs[jt & 1];

        // S strip = Q(16x64) x K^T(64x64): 4 tiles x 2 wmma
        const int kb = hiH ? 16 : 0;
        v8f sfr[4];
#pragma unroll
        for (int t = 0; t < 4; ++t) {
            const half_t* kr = &Kb[(t * 16 + lr) * SK];
            v16h bk0 = hcat(*(const v8h*)(kr + ko),      *(const v8h*)(kr + 16 + ko));
            v16h bk1 = hcat(*(const v8h*)(kr + 32 + ko), *(const v8h*)(kr + 48 + ko));
            v8f z = {};
            z = WMMA_F16(aq0, bk0, z);
            z = WMMA_F16(aq1, bk1, z);
            sfr[t] = z;
        }

        // causal mask + online softmax
        const int keybase = jt * 64;
#pragma unroll
        for (int j = 0; j < 8; ++j) {
            const int q = qtile * 64 + wave * 16 + j + hiH * 8;
            float rmax = -1e30f;
#pragma unroll
            for (int t = 0; t < 4; ++t) {
                const int key = keybase + t * 16 + lr;
                float v = (key <= q) ? sfr[t][j] : -1e30f;
                sfr[t][j] = v;
                rmax = fmaxf(rmax, v);
            }
            rmax = hw_max16(rmax);
            const float mnew  = fmaxf(m_i[j], rmax);
            const float alpha = __expf(m_i[j] - mnew);
            float rsum = 0.0f;
#pragma unroll
            for (int t = 0; t < 4; ++t) {
                float p = __expf(sfr[t][j] - mnew);
                sfr[t][j] = p;
                rsum += p;
            }
            rsum = hw_sum16(rsum);
            l_i[j] = l_i[j] * alpha + rsum;
            m_i[j] = mnew;
#pragma unroll
            for (int t = 0; t < 4; ++t) o[t][j] *= alpha;
        }

        // C-layout P -> LDS -> A-layout fragments
        half_t* pw = &Ps[wave][0];
#pragma unroll
        for (int j = 0; j < 8; ++j) {
            const int M = j + hiH * 8;
#pragma unroll
            for (int t = 0; t < 4; ++t)
                pw[M * SK + t * 16 + lr] = (half_t)sfr[t][j];
        }
        __syncthreads();   // uniform across waves (same trip count)

        const half_t* pr = &Ps[wave][lr * SK];
        v16h ap0 = hcat(*(const v8h*)(pr + ko),      *(const v8h*)(pr + 16 + ko));
        v16h ap1 = hcat(*(const v8h*)(pr + 32 + ko), *(const v8h*)(pr + 48 + ko));
#pragma unroll
        for (int t = 0; t < 4; ++t) {
            const half_t* vr = &Vb[(t * 16 + lr) * SK];
            v16h bv0 = hcat(*(const v8h*)(vr + ko),      *(const v8h*)(vr + 16 + ko));
            v16h bv1 = hcat(*(const v8h*)(vr + 32 + ko), *(const v8h*)(vr + 48 + ko));
            o[t] = WMMA_F16(ap0, bv0, o[t]);
            o[t] = WMMA_F16(ap1, bv1, o[t]);
        }
    }

    // normalize and write attn-out as f16 (B,T,C)
    const int b = bh >> 4, h = bh & 15;
#pragma unroll
    for (int t = 0; t < 4; ++t) {
#pragma unroll
        for (int j = 0; j < 8; ++j) {
            const int q = qtile * 64 + wave * 16 + j + hiH * 8;
            size_t off = ((size_t)b * TT + q) * CC + h * HD + t * 16 + lr;
            Ah[off] = (half_t)(o[t][j] / l_i[j]);
        }
    }
}

// ============================================================================
// Kernel 3: output projection.  Out(4096 x 1024) = Ah(4096 x 1024) @ Wouth^T.
// Same double-buffered all-async structure as kernel 1; fp32 epilogue.
// ============================================================================
__global__ __launch_bounds__(128)
void out_gemm_kernel(const half_t* __restrict__ A, const half_t* __restrict__ Wh,
                     float* __restrict__ Out)
{
    __shared__ half_t As[2][128 * SA];
    __shared__ half_t Bs[2][64 * SA];
    const int tid  = threadIdx.x;
    const int wave = tid >> 5;
    const int lane = tid & 31;
    const int hiH  = (lane >> 4) & 1;
    const int lr   = lane & 15;
    const int m0   = blockIdx.y * 128;
    const int n0   = blockIdx.x * 64;

    auto stage = [&](int k0, int buf) {
        const uint64_t abase = (uint64_t)(uintptr_t)(A + (size_t)m0 * CC + k0);
#pragma unroll
        for (int i = 0; i < 4; ++i) {
            const int lin = ((wave * 4 + i) * 32 + lane) * 8;
            const int r = lin >> 5, c = lin & 31;
            async_b128((uint32_t)(uintptr_t)&As[buf][r * SA + c],
                       (uint32_t)((r * CC + c) * 2), abase);
        }
        const uint64_t bbase = (uint64_t)(uintptr_t)(Wh + (size_t)n0 * CC + k0);
#pragma unroll
        for (int i = 0; i < 2; ++i) {
            const int lin = ((wave * 2 + i) * 32 + lane) * 8;
            const int r = lin >> 5, c = lin & 31;
            async_b128((uint32_t)(uintptr_t)&Bs[buf][r * SA + c],
                       (uint32_t)((r * CC + c) * 2), bbase);
        }
    };

    v8f acc[2][4] = {};
    stage(0, 0);

    for (int ki = 0; ki < CC / 32; ++ki) {
        wait_async0();
        __syncthreads();
        if (ki + 1 < CC / 32) stage((ki + 1) * 32, (ki + 1) & 1);

        const half_t* Ab = As[ki & 1];
        const half_t* Bb = Bs[ki & 1];
        const int ko = hiH ? 8 : 0;
        const half_t* ar0 = &Ab[(wave * 16 + lr) * SA];
        const half_t* ar1 = &Ab[(64 + wave * 16 + lr) * SA];
        v16h a0 = hcat(*(const v8h*)(ar0 + ko), *(const v8h*)(ar0 + 16 + ko));
        v16h a1 = hcat(*(const v8h*)(ar1 + ko), *(const v8h*)(ar1 + 16 + ko));
        const int kb = hiH ? 16 : 0;
#pragma unroll
        for (int t = 0; t < 4; ++t) {
            const half_t* br = &Bb[(t * 16 + lr) * SA];
            v16h bfrag = hcat(*(const v8h*)(br + ko), *(const v8h*)(br + 16 + ko));
            acc[0][t] = WMMA_F16(a0, bfrag, acc[0][t]);
            acc[1][t] = WMMA_F16(a1, bfrag, acc[1][t]);
        }
    }

#pragma unroll
    for (int t = 0; t < 4; ++t) {
        const int n = n0 + t * 16 + lr;
#pragma unroll
        for (int s = 0; s < 2; ++s) {
#pragma unroll
            for (int j = 0; j < 8; ++j) {
                const int m = m0 + s * 64 + wave * 16 + j + hiH * 8;
                Out[(size_t)m * CC + n] = acc[s][t][j];
            }
        }
    }
}

// ============================================================================
// Host launcher
// ============================================================================
extern "C" void kernel_launch(void* const* d_in, const int* in_sizes, int n_in,
                              void* d_out, int out_size, void* d_ws, size_t ws_size,
                              hipStream_t stream)
{
    const float* x     = (const float*)d_in[0];   // (2,2048,1024) f32
    const float* W_qkv = (const float*)d_in[1];   // (3072,1024)   f32
    const float* W_out = (const float*)d_in[2];   // (1024,1024)   f32
    float* out = (float*)d_out;                   // (2,2048,1024) f32

    const size_t nX  = (size_t)BB * TT * CC;      // 4,194,304
    const size_t nWq = (size_t)3 * CC * CC;       // 3,145,728
    const size_t nWo = (size_t)CC * CC;           // 1,048,576
    const size_t qkvElems = (size_t)BB * HH * TT * HD;   // 4,194,304

    char* ws = (char*)d_ws;
    half_t* Qh  = (half_t*)(ws);                              // 8 MiB
    half_t* Kh  = Qh + qkvElems;                              // 8 MiB
    half_t* Vt  = Kh + qkvElems;                              // 8 MiB (B,H,hd,T)
    half_t* Ah  = Vt + qkvElems;                              // 8 MiB
    half_t* Xh  = Ah + qkvElems;                              // 8 MiB
    half_t* Wqh = Xh + nX;                                    // 6 MiB
    half_t* Woh = Wqh + nWq;                                  // 2 MiB

    // 0) one-shot f32 -> f16 conversions
    cvt_f16_kernel<<<(int)(nX  / 8 + 255) / 256, 256, 0, stream>>>(x,     Xh,  (int)(nX  / 8));
    cvt_f16_kernel<<<(int)(nWq / 8 + 255) / 256, 256, 0, stream>>>(W_qkv, Wqh, (int)(nWq / 8));
    cvt_f16_kernel<<<(int)(nWo / 8 + 255) / 256, 256, 0, stream>>>(W_out, Woh, (int)(nWo / 8));

    {   // 1) QKV projection: M=4096, N=3072
        dim3 grid(3 * CC / 64, BB * TT / 128);   // (48, 32)
        qkv_gemm_kernel<<<grid, 128, 0, stream>>>(Xh, Wqh, Qh, Kh, Vt);
    }
    {   // 2) attention: 32 q-tiles x 32 (b,h)
        dim3 grid(TT / 64, BB * HH);             // (32, 32)
        attn_kernel<<<grid, 128, 0, stream>>>(Qh, Kh, Vt, Ah);
    }
    {   // 3) output projection: M=4096, N=1024
        dim3 grid(CC / 64, BB * TT / 128);       // (16, 32)
        out_gemm_kernel<<<grid, 128, 0, stream>>>(Ah, Woh, out);
    }
}
